// MPNN_33698313404933
// MI455X (gfx1250) — matrix operand order repaired and checked by reading, not compiled, not run on hardware
//
#include <hip/hip_runtime.h>

typedef _Float16 half_t;
typedef __attribute__((ext_vector_type(16))) _Float16 v16h;
typedef __attribute__((ext_vector_type(8)))  float    v8f;
typedef __attribute__((ext_vector_type(4)))  unsigned u32x4;

#define BATCH 48
#define NN    32
#define NODES (BATCH*NN)      // 1536
#define EDGES (BATCH*NN*NN)   // 49152
#define HSS   64

union Frag { u32x4 q[2]; v16h v; };

// ---------------------------------------------------------------------------
// Generic f16 GEMM with WMMA:  C[M,N] = act(A[M,K] @ B[K,N] + bias)
// A row-major [M,K] (16B-aligned rows, K%32==0).  B packed in WMMA fragment
// order: Bp[((k0/32)*Nstride + n)*32 + kk]  (Nstride = N rounded up to 64).
// Block = 128 threads = 4 waves; wave w computes rows [w*16,w*16+16) x 64 cols.
// All fragments are 2x 128-bit global loads (no LDS, no barriers).
// ---------------------------------------------------------------------------
__global__ void __launch_bounds__(128)
gemm_wmma(const half_t* __restrict__ A, const half_t* __restrict__ Bp,
          const float* __restrict__ bias, int Nb,
          half_t* __restrict__ outH, float* __restrict__ outF,
          int M, int N, int Nstride, int K, int act)
{
    const int tid  = threadIdx.x;
    const int lane = tid & 31;
    const int wid  = tid >> 5;
    const int hlane = lane >> 4;
    const int l15   = lane & 15;
    const int rowTile = blockIdx.y * 64;
    const int colTile = blockIdx.x * 64;

    const int row = rowTile + wid * 16 + l15;
    const half_t* Arow = A + (size_t)row * K;

    v8f acc[4];
    for (int t = 0; t < 4; ++t)
        for (int j = 0; j < 8; ++j) acc[t][j] = 0.0f;

    for (int k0 = 0; k0 < K; k0 += 32) {
        // A fragment: lane(h=0) holds K 0-15; lane(h=1) K 16-31
        Frag fa;
        fa.q[0] = *(const u32x4*)(Arow + k0 + hlane * 16);
        fa.q[1] = *(const u32x4*)(Arow + k0 + hlane * 16 + 8);
        const half_t* Bk = Bp + (size_t)(k0 >> 5) * Nstride * 32;
        for (int t = 0; t < 4; ++t) {
            int col = colTile + t * 16 + l15;
            const half_t* bptr = Bk + (size_t)col * 32 + hlane * 16;
            Frag fb;
            fb.q[0] = *(const u32x4*)(bptr);
            fb.q[1] = *(const u32x4*)(bptr + 8);
            acc[t] = __builtin_amdgcn_wmma_f32_16x16x32_f16(
                false, fa.v, false, fb.v, (short)0, acc[t], false, false);
        }
    }

    for (int t = 0; t < 4; ++t) {
        int col = colTile + t * 16 + l15;
        if (col >= N) continue;
        float bv = (bias != nullptr && col < Nb) ? bias[col] : 0.0f;
        for (int r = 0; r < 8; ++r) {
            int orow = rowTile + wid * 16 + ((lane < 16) ? r : (r + 8));
            float v = acc[t][r] + bv;
            if (act == 1) v = v > 0.0f ? v : 0.0f;
            size_t o = (size_t)orow * N + col;
            if (outH) outH[o] = (half_t)v;
            if (outF) outF[o] = v;
        }
    }
}

// ---------------------------------------------------------------------------
// Message kernel: one block (64 thr = 2 waves) per (b,v).
// Computes m^T[o,w] = Ut[b,v][o,:] @ z[b,v,w,:]^T + Bh, then scatters
// g-gated values into m_f[b,w,o] and m_types[k,b,w,o] with f32 atomics.
// Ut row-major [o=64][c=128]; z row-major [w=32][c=128]: both fragment paths
// are contiguous 128-bit loads.
// ---------------------------------------------------------------------------
__global__ void __launch_bounds__(64)
msg_wmma(const half_t* __restrict__ z, const half_t* __restrict__ Ut,
         const float* __restrict__ Bh, const float* __restrict__ g,
         const float* __restrict__ bonehot,
         float* __restrict__ m_f, float* __restrict__ m_types)
{
    __shared__ int ktype[32];
    const int node = blockIdx.x;          // b*32 + v
    const int b = node >> 5, v = node & 31;
    const half_t* Az = z  + (size_t)node * 32 * 128;   // [w][c]
    const half_t* Au = Ut + (size_t)node * 8192;       // [o][c]
    const int tid = threadIdx.x;
    const int lane = tid & 31, wid = tid >> 5;
    const int hlane = lane >> 4, l15 = lane & 15;

    if (tid < 32) {
        const float* bo = bonehot + ((size_t)b * 1024 + v * 32 + tid) * 5;
        int kk = 0; float best = bo[0];
        for (int k = 1; k < 5; ++k) if (bo[k] > best) { best = bo[k]; kk = k; }
        ktype[tid] = kk;
    }
    __syncthreads();

    v8f acc[2][2];
    for (int mt = 0; mt < 2; ++mt)
        for (int nt = 0; nt < 2; ++nt)
            for (int j = 0; j < 8; ++j) acc[mt][nt][j] = 0.0f;

    for (int k0 = 0; k0 < 128; k0 += 32) {
        Frag fa[2];
        for (int mt = 0; mt < 2; ++mt) {
            int o = (wid * 2 + mt) * 16 + l15;
            const half_t* base = Au + (size_t)o * 128 + k0;
            fa[mt].q[0] = *(const u32x4*)(base + hlane * 16);
            fa[mt].q[1] = *(const u32x4*)(base + hlane * 16 + 8);
        }
        Frag fb[2];
        for (int nt = 0; nt < 2; ++nt) {
            int w = nt * 16 + l15;
            const half_t* base = Az + (size_t)w * 128 + k0;
            fb[nt].q[0] = *(const u32x4*)(base + hlane * 16);
            fb[nt].q[1] = *(const u32x4*)(base + hlane * 16 + 8);
        }
        for (int mt = 0; mt < 2; ++mt)
            for (int nt = 0; nt < 2; ++nt)
                acc[mt][nt] = __builtin_amdgcn_wmma_f32_16x16x32_f16(
                    false, fa[mt].v, false, fb[nt].v, (short)0, acc[mt][nt], false, false);
    }

    for (int mt = 0; mt < 2; ++mt) {
        for (int nt = 0; nt < 2; ++nt) {
            int w = nt * 16 + l15;
            float gv = g[(size_t)b * 1024 + v * 32 + w];
            if (gv == 0.0f) continue;
            int kt = ktype[w];
            for (int r = 0; r < 8; ++r) {
                int o = (wid * 2 + mt) * 16 + ((lane < 16) ? r : (r + 8));
                float val = (acc[mt][nt][r] + Bh[(size_t)node * 64 + o]) * gv;
                atomicAdd(&m_f[((size_t)(b * 32 + w)) * 64 + o], val);
                atomicAdd(&m_types[(((size_t)kt * NODES) + b * 32 + w) * 64 + o], val);
            }
        }
    }
}

// ---------------------------------------------------------------------------
// Small helper kernels
// ---------------------------------------------------------------------------
__global__ void pack_e_kernel(const float* __restrict__ e, half_t* __restrict__ eh)
{
    int i = blockIdx.x * 256 + threadIdx.x;
    if (i >= EDGES * 32) return;
    int edge = i >> 5, c = i & 31;
    eh[i] = (c < 8) ? (half_t)e[edge * 8 + c] : (half_t)0.0f;
}

// W [outd, ind] f32 -> fragment-ordered f16: out[((k/32)*Np64 + n)*32 + (k%32)]
__global__ void pack_wt_kernel(const float* __restrict__ W, half_t* __restrict__ out,
                               int kin, int nout, int Kp, int Np64)
{
    int i = blockIdx.x * 256 + threadIdx.x;
    if (i >= Kp * Np64) return;
    int slice = Np64 * 32;
    int ks = i / slice, rem = i % slice;
    int n = rem >> 5, kk = rem & 31;
    int k = ks * 32 + kk;
    out[i] = (k < kin && n < nout) ? (half_t)W[(size_t)n * kin + k] : (half_t)0.0f;
}

// W4 [4096,128] -> fragment-ordered Uw for the U-GEMM with column n = o*128 + c,
// K = i (64):  Uw[((i/32)*8192 + o*128 + c)*32 + (i%32)] = W4[(o*64+i)*128 + c]
__global__ void pack_U_kernel(const float* __restrict__ W4, half_t* __restrict__ Uw)
{
    int i = blockIdx.x * 256 + threadIdx.x;
    if (i >= 64 * 8192) return;
    int ks = i >> 18;                 // / (8192*32)
    int rem = i & ((8192 * 32) - 1);
    int n = rem >> 5, kk = rem & 31;
    int ii = ks * 32 + kk;            // K index (= i of W4 inner dim)
    int o = n >> 7, c = n & 127;
    Uw[i] = (half_t)W4[((size_t)o * 64 + ii) * 128 + c];
}

__global__ void init_h_kernel(const float* __restrict__ h_in,
                              float* h0F, half_t* h0H, float* hF, half_t* hH,
                              float* mask)
{
    int node = blockIdx.x, j = threadIdx.x;  // 64 threads
    float v = (j < 16) ? h_in[(size_t)node * 16 + j] : 0.0f;
    size_t o = (size_t)node * 64 + j;
    h0F[o] = v; h0H[o] = (half_t)v; hF[o] = v; hH[o] = (half_t)v;
    if (j == 0) {
        float s = 0.0f;
        for (int t = 0; t < 16; ++t) s += h_in[(size_t)node * 16 + t];
        mask[node] = (s > 0.0f) ? 1.0f : 0.0f;
    }
}

__global__ void bh_kernel(const float* __restrict__ b4, const float* __restrict__ hF,
                          float* __restrict__ Bh)
{
    int node = blockIdx.x, o = threadIdx.x;  // 64 threads
    const float* h = hF + (size_t)node * 64;
    const float* w = b4 + (size_t)o * 64;
    float s = 0.0f;
    for (int i = 0; i < 64; ++i) s += w[i] * h[i];
    Bh[(size_t)node * 64 + o] = s;
}

__global__ void f2h_kernel(const float* __restrict__ in, half_t* __restrict__ out, int n)
{
    int i = blockIdx.x * 256 + threadIdx.x;
    if (i < n) out[i] = (half_t)in[i];
}

__device__ __forceinline__ float sigf(float x) { return 1.0f / (1.0f + __expf(-x)); }

__global__ void gru_combine_kernel(const float* __restrict__ gi0, const float* __restrict__ gh0,
                                   const float* __restrict__ gib, const float* __restrict__ ghb,
                                   const float* __restrict__ mask, const float* __restrict__ vaPtr,
                                   float* __restrict__ hF, half_t* __restrict__ hH)
{
    int node = blockIdx.x, j = threadIdx.x;  // 64 threads
    size_t o = (size_t)node * 64 + j;
    float h = hF[o];
    const float* gi = gi0 + (size_t)node * 192;
    const float* gh = gh0 + (size_t)node * 192;
    float r  = sigf(gi[j]       + gh[j]);
    float zz = sigf(gi[64 + j]  + gh[64 + j]);
    float n  = tanhf(gi[128 + j] + r * gh[128 + j]);
    float h_f = (1.0f - zz) * n + zz * h;

    const float* ghB = ghb + (size_t)node * 192;
    float h_tb = 0.0f;
    for (int k = 0; k < 5; ++k) {
        const float* gk = gib + ((size_t)k * NODES + node) * 192;
        float rk = sigf(gk[j]      + ghB[j]);
        float zk = sigf(gk[64 + j] + ghB[64 + j]);
        float nk = tanhf(gk[128 + j] + rk * ghB[128 + j]);
        h_tb += (1.0f - zk) * nk + zk * h;
    }
    float va = sigf(vaPtr[0]);
    float hn = mask[node] * (va * h_f + (1.0f - va) * h_tb);
    hF[o] = hn;
    hH[o] = (half_t)hn;
}

__global__ void cat_kernel(const half_t* __restrict__ h0H, const half_t* __restrict__ hH,
                           half_t* __restrict__ catH)
{
    int node = blockIdx.x, j = threadIdx.x;  // 128 threads
    catH[(size_t)node * 128 + j] =
        (j < 64) ? h0H[(size_t)node * 64 + j] : hH[(size_t)node * 64 + (j - 64)];
}

__global__ void readout_sum_kernel(const float* __restrict__ gate, const float* __restrict__ jv,
                                   int D, float* __restrict__ out)
{
    int b = blockIdx.x, d = threadIdx.x;  // 32 threads
    if (d >= D) return;
    float s = 0.0f;
    for (int n = 0; n < 32; ++n) {
        size_t idx = ((size_t)(b * 32 + n)) * 16 + d;
        s += sigf(gate[idx]) * jv[idx];
    }
    out[(size_t)b * D + d] = s;
}

// ---------------------------------------------------------------------------
// Host orchestration
// ---------------------------------------------------------------------------
static inline void* bump(char*& p, size_t bytes)
{
    void* r = (void*)p;
    p += (bytes + 255) & ~(size_t)255;
    return r;
}

static inline void launch_gemm(const half_t* A, const half_t* B, const float* bias, int Nb,
                               half_t* outH, float* outF, int M, int N, int Nstride, int K,
                               int act, hipStream_t stream)
{
    dim3 grid((unsigned)((N + 63) / 64), (unsigned)(M / 64));
    gemm_wmma<<<grid, 128, 0, stream>>>(A, B, bias, Nb, outH, outF, M, N, Nstride, K, act);
}

extern "C" void kernel_launch(void* const* d_in, const int* in_sizes, int n_in,
                              void* d_out, int out_size, void* d_ws, size_t ws_size,
                              hipStream_t stream)
{
    (void)in_sizes; (void)n_in; (void)out_size; (void)ws_size;

    const float* g    = (const float*)d_in[0];
    const float* h_in = (const float*)d_in[1];
    const float* e    = (const float*)d_in[2];
    const float* bo   = (const float*)d_in[3];
    const float* va   = (const float*)d_in[4];
    const float* eW[4] = {(const float*)d_in[5], (const float*)d_in[7],
                          (const float*)d_in[9], (const float*)d_in[11]};
    const float* eB[4] = {(const float*)d_in[6], (const float*)d_in[8],
                          (const float*)d_in[10], (const float*)d_in[12]};
    const float* Wih0 = (const float*)d_in[13];
    const float* Whh0 = (const float*)d_in[14];
    const float* bih0 = (const float*)d_in[15];
    const float* bhh0 = (const float*)d_in[16];
    const float* Wihb = (const float*)d_in[17];
    const float* Whhb = (const float*)d_in[18];
    const float* bihb = (const float*)d_in[19];
    const float* bhhb = (const float*)d_in[20];
    float* out = (float*)d_out;

    // -------- workspace layout --------
    char* p = (char*)d_ws;
    half_t* eh  = (half_t*)bump(p, (size_t)EDGES * 32 * 2);
    half_t* t1  = (half_t*)bump(p, (size_t)EDGES * 128 * 2);  // reused: GRU preacts
    half_t* t2  = (half_t*)bump(p, (size_t)EDGES * 256 * 2);  // reused: Ut
    half_t* zb  = (half_t*)bump(p, (size_t)EDGES * 128 * 2);
    half_t* W1t = (half_t*)bump(p, (size_t)32 * 128 * 2);
    half_t* W2t = (half_t*)bump(p, (size_t)128 * 256 * 2);
    half_t* W3t = (half_t*)bump(p, (size_t)256 * 128 * 2);
    half_t* Uw  = (half_t*)bump(p, (size_t)64 * 8192 * 2);
    half_t* Wih0t = (half_t*)bump(p, (size_t)64 * 192 * 2);
    half_t* Whh0t = (half_t*)bump(p, (size_t)64 * 192 * 2);
    half_t* Wihbt = (half_t*)bump(p, (size_t)64 * 192 * 2);
    half_t* Whhbt = (half_t*)bump(p, (size_t)64 * 192 * 2);
    half_t* rpool = (half_t*)bump(p, (size_t)(4 * 90112 + 4 * 81920) * 2);
    float*  h0F  = (float*)bump(p, (size_t)NODES * 64 * 4);
    half_t* h0H  = (half_t*)bump(p, (size_t)NODES * 64 * 2);
    float*  hF   = (float*)bump(p, (size_t)NODES * 64 * 4);
    half_t* hH   = (half_t*)bump(p, (size_t)NODES * 64 * 2);
    float*  mask = (float*)bump(p, (size_t)NODES * 4);
    float*  Bh   = (float*)bump(p, (size_t)NODES * 64 * 4);
    float*  m_fF = (float*)bump(p, (size_t)NODES * 64 * 4);
    half_t* m_fH = (half_t*)bump(p, (size_t)NODES * 64 * 2);
    float*  m_tF = (float*)bump(p, (size_t)5 * NODES * 64 * 4);
    half_t* m_tH = (half_t*)bump(p, (size_t)5 * NODES * 64 * 2);
    half_t* catH = (half_t*)bump(p, (size_t)NODES * 128 * 2);
    half_t* rA   = (half_t*)bump(p, (size_t)NODES * 256 * 2);
    half_t* rB   = (half_t*)bump(p, (size_t)NODES * 256 * 2);
    float*  gateF = (float*)bump(p, (size_t)NODES * 16 * 4);
    float*  jF    = (float*)bump(p, (size_t)NODES * 16 * 4);

    half_t* Ut = t2;                // 1536*8192 f16 == EDGES*256 f16
    float* gi0 = (float*)t1;        // GRU preact pool inside t1
    float* gh0 = gi0 + (size_t)NODES * 192;
    float* ghb = gh0 + (size_t)NODES * 192;
    float* gib = ghb + (size_t)NODES * 192;   // [5*NODES, 192]

    // -------- pack (every call; no cached state) --------
    pack_e_kernel<<<(EDGES * 32 + 255) / 256, 256, 0, stream>>>(e, eh);
    pack_wt_kernel<<<(32 * 128 + 255) / 256, 256, 0, stream>>>(eW[0], W1t, 8, 128, 32, 128);
    pack_wt_kernel<<<(128 * 256 + 255) / 256, 256, 0, stream>>>(eW[1], W2t, 128, 256, 128, 256);
    pack_wt_kernel<<<(256 * 128 + 255) / 256, 256, 0, stream>>>(eW[2], W3t, 256, 128, 256, 128);
    pack_U_kernel<<<(64 * 8192 + 255) / 256, 256, 0, stream>>>(eW[3], Uw);
    pack_wt_kernel<<<(64 * 192 + 255) / 256, 256, 0, stream>>>(Wih0, Wih0t, 64, 192, 64, 192);
    pack_wt_kernel<<<(64 * 192 + 255) / 256, 256, 0, stream>>>(Whh0, Whh0t, 64, 192, 64, 192);
    pack_wt_kernel<<<(64 * 192 + 255) / 256, 256, 0, stream>>>(Wihb, Wihbt, 64, 192, 64, 192);
    pack_wt_kernel<<<(64 * 192 + 255) / 256, 256, 0, stream>>>(Whhb, Whhbt, 64, 192, 64, 192);

    // readout MLP packing: i-MLPs (in=128) and j-MLPs (in=64), 4 layers each
    const int iBase[4] = {21, 37, 53, 69};
    const int jBase[4] = {29, 45, 61, 77};
    const int Dp[4]    = {1, 7, 7, 1};
    const int outOff[4] = {0, 48, 384, 720};
    const int iK[4] = {128, 128, 256, 128}, jK[4] = {64, 128, 256, 128};
    const int Lnout[4] = {128, 256, 128, 0};   // true out (last filled per-readout)
    const int Lp64[4]  = {128, 256, 128, 64};  // 64-padded N for packing
    half_t* WtI[4][4];
    half_t* WtJ[4][4];
    const float* BiI[4][4];
    const float* BiJ[4][4];
    {
        size_t off = 0;
        for (int pidx = 0; pidx < 4; ++pidx) {
            for (int l = 0; l < 4; ++l) {
                int kin = iK[l];
                int nout = (l < 3) ? Lnout[l] : Dp[pidx];
                WtI[pidx][l] = rpool + off;
                BiI[pidx][l] = (const float*)d_in[iBase[pidx] + 2 * l + 1];
                pack_wt_kernel<<<(kin * Lp64[l] + 255) / 256, 256, 0, stream>>>(
                    (const float*)d_in[iBase[pidx] + 2 * l], WtI[pidx][l],
                    kin, nout, kin, Lp64[l]);
                off += (size_t)kin * Lp64[l];
            }
            for (int l = 0; l < 4; ++l) {
                int kin = jK[l];
                int nout = (l < 3) ? Lnout[l] : Dp[pidx];
                WtJ[pidx][l] = rpool + off;
                BiJ[pidx][l] = (const float*)d_in[jBase[pidx] + 2 * l + 1];
                pack_wt_kernel<<<(kin * Lp64[l] + 255) / 256, 256, 0, stream>>>(
                    (const float*)d_in[jBase[pidx] + 2 * l], WtJ[pidx][l],
                    kin, nout, kin, Lp64[l]);
                off += (size_t)kin * Lp64[l];
            }
        }
    }

    init_h_kernel<<<NODES, 64, 0, stream>>>(h_in, h0F, h0H, hF, hH, mask);

    // -------- edge feature MLP (z = relu chain, layer-invariant) --------
    launch_gemm(eh, W1t, eB[0], 128, t1, nullptr, EDGES, 128, 128, 32, 1, stream);
    launch_gemm(t1, W2t, eB[1], 256, t2, nullptr, EDGES, 256, 256, 128, 1, stream);
    launch_gemm(t2, W3t, eB[2], 128, zb, nullptr, EDGES, 128, 128, 256, 1, stream);
    // t1 (preacts) and t2 (Ut) now free for reuse

    // -------- message-passing layers --------
    for (int layer = 0; layer < 3; ++layer) {
        // Ut[b,v][o][c] = h_t contracted with reshaped W4 (M=1536, K=64, N=8192)
        launch_gemm(hH, Uw, nullptr, 0, Ut, nullptr, NODES, 8192, 8192, 64, 0, stream);
        bh_kernel<<<NODES, 64, 0, stream>>>(eB[3], hF, Bh);
        hipMemsetAsync(m_fF, 0, (size_t)NODES * 64 * 4, stream);
        hipMemsetAsync(m_tF, 0, (size_t)5 * NODES * 64 * 4, stream);
        msg_wmma<<<NODES, 64, 0, stream>>>(zb, Ut, Bh, g, bo, m_fF, m_tF);
        f2h_kernel<<<(NODES * 64 + 255) / 256, 256, 0, stream>>>(m_fF, m_fH, NODES * 64);
        f2h_kernel<<<(5 * NODES * 64 + 255) / 256, 256, 0, stream>>>(m_tF, m_tH, 5 * NODES * 64);
        // GRU pre-activations via WMMA GEMM
        launch_gemm(m_fH, Wih0t, bih0, 192, nullptr, gi0, NODES, 192, 192, 64, 0, stream);
        launch_gemm(hH,   Whh0t, bhh0, 192, nullptr, gh0, NODES, 192, 192, 64, 0, stream);
        launch_gemm(m_tH, Wihbt, bihb, 192, nullptr, gib, 5 * NODES, 192, 192, 64, 0, stream);
        launch_gemm(hH,   Whhbt, bhhb, 192, nullptr, ghb, NODES, 192, 192, 64, 0, stream);
        gru_combine_kernel<<<NODES, 64, 0, stream>>>(gi0, gh0, gib, ghb, mask, va, hF, hH);
    }

    // -------- readouts --------
    cat_kernel<<<NODES, 128, 0, stream>>>(h0H, hH, catH);
    for (int pidx = 0; pidx < 4; ++pidx) {
        // gate = mlp_i([h0, hT])
        launch_gemm(catH, WtI[pidx][0], BiI[pidx][0], 128, rA, nullptr, NODES, 128, 128, 128, 1, stream);
        launch_gemm(rA,   WtI[pidx][1], BiI[pidx][1], 256, rB, nullptr, NODES, 256, 256, 128, 1, stream);
        launch_gemm(rB,   WtI[pidx][2], BiI[pidx][2], 128, rA, nullptr, NODES, 128, 128, 256, 1, stream);
        launch_gemm(rA,   WtI[pidx][3], BiI[pidx][3], Dp[pidx], nullptr, gateF, NODES, 16, 64, 128, 0, stream);
        // j = mlp_j(hT)
        launch_gemm(hH,   WtJ[pidx][0], BiJ[pidx][0], 128, rA, nullptr, NODES, 128, 128, 64, 1, stream);
        launch_gemm(rA,   WtJ[pidx][1], BiJ[pidx][1], 256, rB, nullptr, NODES, 256, 256, 128, 1, stream);
        launch_gemm(rB,   WtJ[pidx][2], BiJ[pidx][2], 128, rA, nullptr, NODES, 128, 128, 256, 1, stream);
        launch_gemm(rA,   WtJ[pidx][3], BiJ[pidx][3], Dp[pidx], nullptr, jF, NODES, 16, 64, 128, 0, stream);
        readout_sum_kernel<<<BATCH, 32, 0, stream>>>(gateF, jF, Dp[pidx], out + outOff[pidx]);
    }
}
